// implicit_sdf_57286273794209
// MI455X (gfx1250) — matrix operand order repaired and checked by reading, not compiled, not run on hardware
//
#include <hip/hip_runtime.h>
#include <hip/hip_bf16.h>
#include <math.h>

typedef __attribute__((ext_vector_type(16))) _Float16 v16h;
typedef __attribute__((ext_vector_type(8)))  float    v8f;

// ---------------------------------------------------------------------------
// Workspace layout (bytes)
// ---------------------------------------------------------------------------
#define OFF_B0   0          // 16*16*64 f16 = 32768 B
#define OFF_B1   32768      // 16*16*32 f16 = 16384 B
#define OFF_B2   49152
#define OFF_B3   65536
#define OFF_W1T  81920      // 112*32  f16 = 7168 B
#define OFF_W2T  89088      // 112*128 f16 = 28672 B
#define OFF_B1P  117760     // 112 f32
#define OFF_B2P  118208     // 112 f32
#define OFF_W3T  118784     // 16*128 f16 = 4096 B
#define OFF_PE   131072     // 32768*64 f16 = 4 MB posenc table

// ---------------------------------------------------------------------------
// WMMA helper: D = A(16x32 f16) * B(32x16 f16) + C(16x16 f32)
// ---------------------------------------------------------------------------
__device__ __forceinline__ v8f wmma_f16(v16h a, v16h b, v8f c) {
  return __builtin_amdgcn_wmma_f32_16x16x32_f16(
      false, a, false, b, (short)0, c, false, false);
}

// A-fragment from a row-major [16][kpad] f16 tile (LDS or global).
// ISA 16-bit A 16x32: lane L row M=L%16; half=L/16;
// VGPR j(0..3): K=half*8+2j(+bit); VGPR j(4..7): K=16+half*8+2(j-4)(+bit).
__device__ __forceinline__ v16h load_fragA(const _Float16* src, int kpad, int ks, int lane) {
  const int r  = lane & 15;
  const int hf = lane >> 4;
  union { v16h h; unsigned int u[8]; } out;
#pragma unroll
  for (int j = 0; j < 8; ++j) {
    int kk = ((j < 4) ? 0 : 16) + hf * 8 + ((j & 3) << 1);
    out.u[j] = *(const unsigned int*)(src + r * kpad + ks * 32 + kk);
  }
  return out.h;
}

// B-fragment from row-major [n][kpad] f16 weights (LDS).
// ISA 16-bit B 32x16: lane L col N=L%16; element e<8 holds K=(L/16)*8+e, e>=8 holds K=16+(L/16)*8+(e-8).
__device__ __forceinline__ v16h load_fragB(const _Float16* g, int kpad, int ks, int lane) {
  const int n  = lane & 15;
  const int hf = lane >> 4;
  const _Float16* p = g + n * kpad + ks * 32;
  union { v16h h; uint4 q[2]; } out;
  out.q[0] = *(const uint4*)(p + hf * 8);
  out.q[1] = *(const uint4*)(p + 16 + hf * 8);
  return out.h;
}

// Fast SiLU: v_exp_f32 + v_rcp_f32
__device__ __forceinline__ float silu(float v) {
  return v * __builtin_amdgcn_rcpf(1.0f + __expf(-v));
}

// ---------------------------------------------------------------------------
// Prologue 0: posenc table pe[32768][64] f16 (cols 62,63 zero)
// ---------------------------------------------------------------------------
__global__ void prep_posenc(const float* __restrict__ x, _Float16* __restrict__ pe) {
  int i = blockIdx.x * 256 + threadIdx.x;     // over 32768*64
  int p = i >> 6, c = i & 63;
  float x0 = x[p * 2 + 0], x1 = x[p * 2 + 1];
  float v;
  if (c < 2)       v = (c == 0) ? x0 : x1;
  else if (c < 32) { int t = c - 2;  float xv = (t & 1) ? x1 : x0; v = sinf(xv * (float)(1 << (t >> 1))); }
  else if (c < 62) { int t = c - 32; float xv = (t & 1) ? x1 : x0; v = cosf(xv * (float)(1 << (t >> 1))); }
  else             v = 0.0f;
  pe[i] = (_Float16)v;
}

// ---------------------------------------------------------------------------
// Prologue 1: generate wv weights (einsum + layernorm + pb) as f16 B-matrices
// grid = 4 layers * 16 batch * 16 v = 1024 blocks, 64 threads
// ---------------------------------------------------------------------------
__global__ void prep_wv(const float* __restrict__ latent,
                        const float* pw0, const float* pb0, const float* lnw0, const float* lnb0,
                        const float* pw1, const float* pb1, const float* lnw1, const float* lnb1,
                        const float* pw2, const float* pb2, const float* lnw2, const float* lnb2,
                        const float* pw3, const float* pb3, const float* lnw3, const float* lnb3,
                        _Float16* B0, _Float16* B1, _Float16* B2, _Float16* B3) {
  const int layer = blockIdx.x >> 8;
  const int rem   = blockIdx.x & 255;
  const int b     = rem >> 4;
  const int v     = rem & 15;
  const int ch    = (layer == 0) ? 62 : 16;
  const int kpad  = (layer == 0) ? 64 : 32;

  const float* pw  = (layer == 0) ? pw0  : (layer == 1) ? pw1  : (layer == 2) ? pw2  : pw3;
  const float* pb  = (layer == 0) ? pb0  : (layer == 1) ? pb1  : (layer == 2) ? pb2  : pb3;
  const float* lnw = (layer == 0) ? lnw0 : (layer == 1) ? lnw1 : (layer == 2) ? lnw2 : lnw3;
  const float* lnb = (layer == 0) ? lnb0 : (layer == 1) ? lnb1 : (layer == 2) ? lnb2 : lnb3;
  _Float16*    Bd  = (layer == 0) ? B0   : (layer == 1) ? B1   : (layer == 2) ? B2   : B3;

  const int c = threadIdx.x;
  const float* latv = latent + ((size_t)b * 64 + layer * 16 + v) * 64;

  float acc = 0.0f;
  if (c < ch) {
    const float* w = pw + ((size_t)v * ch + c) * 64;
#pragma unroll 8
    for (int d = 0; d < 64; ++d) acc += latv[d] * w[d];
  }
  __shared__ float s[64];
  s[c] = (c < ch) ? acc : 0.0f;
  __syncthreads();

  float mu = 0.0f;
  for (int i = 0; i < ch; ++i) mu += s[i];
  mu /= (float)ch;
  float var = 0.0f;
  for (int i = 0; i < ch; ++i) { float d = s[i] - mu; var += d * d; }
  var /= (float)ch;

  float val = 0.0f;
  if (c < ch) val = (acc - mu) * rsqrtf(var + 1e-5f) * lnw[c] + lnb[c] + pb[v * ch + c];
  if (c < kpad) Bd[((size_t)b * 16 + v) * kpad + c] = (_Float16)((c < ch) ? val : 0.0f);
}

// ---------------------------------------------------------------------------
// Prologue 2: pad/convert static MLP weights to f16 [n][kpad]
// ---------------------------------------------------------------------------
#define N_W1T 3584      // 112*32
#define N_W2T 14336     // 112*128
#define N_W3T 2048      // 16*128 (row 0 = w3, rest zero)
__global__ void prep_mlp(const float* __restrict__ w1, const float* __restrict__ b1,
                         const float* __restrict__ w2, const float* __restrict__ b2,
                         const float* __restrict__ w3,
                         _Float16* w1t, _Float16* w2t, float* b1p, float* b2p,
                         _Float16* w3t) {
  int i = blockIdx.x * 256 + threadIdx.x;
  if (i < N_W1T) {
    int n = i >> 5, k = i & 31;
    w1t[i] = (_Float16)((n < 100 && k < 16) ? w1[n * 16 + k] : 0.0f);
  } else if (i < N_W1T + N_W2T) {
    int j = i - N_W1T;
    int n = j >> 7, k = j & 127;
    w2t[j] = (_Float16)((n < 100 && k < 100) ? w2[n * 100 + k] : 0.0f);
  } else if (i < N_W1T + N_W2T + 112) {
    int n = i - (N_W1T + N_W2T);
    b1p[n] = (n < 100) ? b1[n] : 0.0f;
  } else if (i < N_W1T + N_W2T + 224) {
    int n = i - (N_W1T + N_W2T + 112);
    b2p[n] = (n < 100) ? b2[n] : 0.0f;
  } else if (i < N_W1T + N_W2T + 224 + N_W3T) {
    int j = i - (N_W1T + N_W2T + 224);
    int n = j >> 7, k = j & 127;
    w3t[j] = (_Float16)((n == 0 && k < 100) ? w3[k] : 0.0f);
  }
}

// ---------------------------------------------------------------------------
// Cooperative f16 copy (global -> LDS), 16 bytes per thread-iteration
// ---------------------------------------------------------------------------
__device__ __forceinline__ void copy_f16(_Float16* dst, const _Float16* src,
                                         int n_f16, int tid, int nt) {
  int n16 = n_f16 >> 3;
  for (int i = tid; i < n16; i += nt)
    ((uint4*)dst)[i] = ((const uint4*)src)[i];
}

// ---------------------------------------------------------------------------
// Fused main kernel: 4 waves/block, each wave runs 4 consecutive 16-pt tiles.
// All waves of a block share one batch -> weights staged in LDS once.
// grid = 2048 blocks (128 blocks/batch * 16 batches)
// ---------------------------------------------------------------------------
__global__ __launch_bounds__(128) void sdf_main(
    const _Float16* __restrict__ pe,
    const float* __restrict__ mb0, const float* __restrict__ mb1,
    const float* __restrict__ mb2, const float* __restrict__ mb3,
    const float* __restrict__ b3,
    const _Float16* __restrict__ B0, const _Float16* __restrict__ B1,
    const _Float16* __restrict__ B2, const _Float16* __restrict__ B3,
    const _Float16* __restrict__ w1t, const float* __restrict__ b1p,
    const _Float16* __restrict__ w2t, const float* __restrict__ b2p,
    const _Float16* __restrict__ w3t,
    float* __restrict__ out) {
  // LDS: 4*2048 f16 staging | 22528 f16 weights  = 61440 B
  __shared__ _Float16 smem[4 * 2048 + 22528];

  const int tid  = threadIdx.x;
  const int lane = tid & 31;
  const int wid  = tid >> 5;
  const int b      = blockIdx.x >> 7;           // batch 0..15
  const int blkInB = blockIdx.x & 127;

  _Float16* buf = smem + wid * 2048;
  _Float16* sB0 = smem + 8192;                  // 1024
  _Float16* sB1 = smem + 9216;                  // 512
  _Float16* sB2 = smem + 9728;                  // 512
  _Float16* sB3 = smem + 10240;                 // 512
  _Float16* sW1 = smem + 10752;                 // 3584
  _Float16* sW2 = smem + 14336;                 // 14336
  _Float16* sW3 = smem + 28672;                 // 2048

  // ---- stage all weights for this batch into LDS ---------------------------
  copy_f16(sB0, B0 + (size_t)b * 1024, 1024, tid, 128);
  copy_f16(sB1, B1 + (size_t)b * 512,  512,  tid, 128);
  copy_f16(sB2, B2 + (size_t)b * 512,  512,  tid, 128);
  copy_f16(sB3, B3 + (size_t)b * 512,  512,  tid, 128);
  copy_f16(sW1, w1t, N_W1T, tid, 128);
  copy_f16(sW2, w2t, N_W2T, tid, 128);
  copy_f16(sW3, w3t, N_W3T, tid, 128);
  __syncthreads();

  const int n  = lane & 15;
  const int hf = lane >> 4;

  // ---- loop-invariant per-lane scalars (global, coalesced) -----------------
  const float b3v = b3[0];
  const float mbias0 = mb0[n], mbias1 = mb1[n], mbias2 = mb2[n], mbias3 = mb3[n];
  float bias1[7], bias2[7];
#pragma unroll
  for (int t = 0; t < 7; ++t) { bias1[t] = b1p[t * 16 + n]; bias2[t] = b2p[t * 16 + n]; }

  // ---- loop-invariant B-fragments held in registers -------------------------
  v16h bb00 = load_fragB(sB0, 64, 0, lane);
  v16h bb01 = load_fragB(sB0, 64, 1, lane);
  v16h bbL1 = load_fragB(sB1, 32, 0, lane);
  v16h bbL2 = load_fragB(sB2, 32, 0, lane);
  v16h bbL3 = load_fragB(sB3, 32, 0, lane);

  for (int t4 = 0; t4 < 4; ++t4) {
    const int tile  = (blkInB * 4 + wid) * 4 + t4;  // 0..2047 within batch
    const int pbase = tile * 16;

    // zero the [16][32]-view K-pad (cols 16..31) once per tile
#pragma unroll
    for (int r = 0; r < 8; ++r)
      buf[(r + hf * 8) * 32 + 16 + n] = (_Float16)0.0f;

    // ---- layer 0: A gathered straight from global posenc table -------------
    {
      const _Float16* peT = pe + (size_t)pbase * 64;
      v16h a0 = load_fragA(peT, 64, 0, lane);
      v16h a1 = load_fragA(peT, 64, 1, lane);
      v8f cc = {};
      cc = wmma_f16(a0, bb00, cc);
      cc = wmma_f16(a1, bb01, cc);
#pragma unroll
      for (int r = 0; r < 8; ++r)
        buf[(r + hf * 8) * 32 + n] = (_Float16)silu(cc[r] + mbias0);
    }

    // ---- layers 1..3: 16 -> 16 ----------------------------------------------
#pragma unroll
    for (int L = 0; L < 3; ++L) {
      v16h bb = (L == 0) ? bbL1 : (L == 1) ? bbL2 : bbL3;
      float bias = (L == 0) ? mbias1 : (L == 1) ? mbias2 : mbias3;
      v16h a = load_fragA(buf, 32, 0, lane);
      v8f cc = {};
      cc = wmma_f16(a, bb, cc);
#pragma unroll
      for (int r = 0; r < 8; ++r)
        buf[(r + hf * 8) * 32 + n] = (_Float16)silu(cc[r] + bias);
    }

    // ---- W1: 16 -> 100 (7 n-tiles) ------------------------------------------
    {
      v16h am = load_fragA(buf, 32, 0, lane);
#pragma unroll
      for (int t = 0; t < 7; ++t) {
        v16h bt = load_fragB(sW1 + t * 16 * 32, 32, 0, lane);
        float bias = bias1[t];
        v8f cc = { bias, bias, bias, bias, bias, bias, bias, bias };
        cc = wmma_f16(am, bt, cc);
#pragma unroll
        for (int r = 0; r < 8; ++r)
          buf[(r + hf * 8) * 128 + t * 16 + n] = (_Float16)silu(cc[r]);
      }
#pragma unroll
      for (int r = 0; r < 8; ++r)
        buf[(r + hf * 8) * 128 + 112 + n] = (_Float16)0.0f;   // K-pad 112..127
    }

    // ---- W2: 100 -> 100 (4 K-steps x 7 n-tiles = 28 WMMA) -------------------
    {
      v16h a20 = load_fragA(buf, 128, 0, lane);
      v16h a21 = load_fragA(buf, 128, 1, lane);
      v16h a22 = load_fragA(buf, 128, 2, lane);
      v16h a23 = load_fragA(buf, 128, 3, lane);
#pragma unroll
      for (int t = 0; t < 7; ++t) {
        const _Float16* wb = sW2 + t * 16 * 128;
        float bias = bias2[t];
        v8f cc = { bias, bias, bias, bias, bias, bias, bias, bias };
        cc = wmma_f16(a20, load_fragB(wb, 128, 0, lane), cc);
        cc = wmma_f16(a21, load_fragB(wb, 128, 1, lane), cc);
        cc = wmma_f16(a22, load_fragB(wb, 128, 2, lane), cc);
        cc = wmma_f16(a23, load_fragB(wb, 128, 3, lane), cc);
#pragma unroll
        for (int r = 0; r < 8; ++r)
          buf[(r + hf * 8) * 128 + t * 16 + n] = (_Float16)silu(cc[r]);
      }
    }

    // ---- W3: 100 -> 1 as WMMA against w3t (row 0 = w3) ----------------------
    {
      v16h a30 = load_fragA(buf, 128, 0, lane);
      v16h a31 = load_fragA(buf, 128, 1, lane);
      v16h a32 = load_fragA(buf, 128, 2, lane);
      v16h a33 = load_fragA(buf, 128, 3, lane);
      v8f c3 = { b3v, b3v, b3v, b3v, b3v, b3v, b3v, b3v };
      c3 = wmma_f16(a30, load_fragB(sW3, 128, 0, lane), c3);
      c3 = wmma_f16(a31, load_fragB(sW3, 128, 1, lane), c3);
      c3 = wmma_f16(a32, load_fragB(sW3, 128, 2, lane), c3);
      c3 = wmma_f16(a33, load_fragB(sW3, 128, 3, lane), c3);
      if (n == 0) {            // lanes 0 and 16 hold D column 0 (rows hf*8+r)
#pragma unroll
        for (int r = 0; r < 8; ++r)
          out[(size_t)b * 32768 + pbase + hf * 8 + r] = c3[r];
      }
    }
  }
}

// ---------------------------------------------------------------------------
// Host launcher
// ---------------------------------------------------------------------------
extern "C" void kernel_launch(void* const* d_in, const int* in_sizes, int n_in,
                              void* d_out, int out_size, void* d_ws, size_t ws_size,
                              hipStream_t stream) {
  (void)in_sizes; (void)n_in; (void)out_size; (void)ws_size;

  const float* x      = (const float*)d_in[0];
  const float* latent = (const float*)d_in[1];
  const float* pw0 = (const float*)d_in[2];  const float* pb0 = (const float*)d_in[3];
  const float* lnw0 = (const float*)d_in[4]; const float* lnb0 = (const float*)d_in[5];
  const float* mb0 = (const float*)d_in[6];
  const float* pw1 = (const float*)d_in[7];  const float* pb1 = (const float*)d_in[8];
  const float* lnw1 = (const float*)d_in[9]; const float* lnb1 = (const float*)d_in[10];
  const float* mb1 = (const float*)d_in[11];
  const float* pw2 = (const float*)d_in[12]; const float* pb2 = (const float*)d_in[13];
  const float* lnw2 = (const float*)d_in[14]; const float* lnb2 = (const float*)d_in[15];
  const float* mb2 = (const float*)d_in[16];
  const float* pw3 = (const float*)d_in[17]; const float* pb3 = (const float*)d_in[18];
  const float* lnw3 = (const float*)d_in[19]; const float* lnb3 = (const float*)d_in[20];
  const float* mb3 = (const float*)d_in[21];
  const float* w1 = (const float*)d_in[22]; const float* b1 = (const float*)d_in[23];
  const float* w2 = (const float*)d_in[24]; const float* b2 = (const float*)d_in[25];
  const float* w3 = (const float*)d_in[26]; const float* b3 = (const float*)d_in[27];

  char* ws = (char*)d_ws;
  _Float16* B0  = (_Float16*)(ws + OFF_B0);
  _Float16* B1  = (_Float16*)(ws + OFF_B1);
  _Float16* B2  = (_Float16*)(ws + OFF_B2);
  _Float16* B3  = (_Float16*)(ws + OFF_B3);
  _Float16* w1t = (_Float16*)(ws + OFF_W1T);
  _Float16* w2t = (_Float16*)(ws + OFF_W2T);
  float*    b1p = (float*)(ws + OFF_B1P);
  float*    b2p = (float*)(ws + OFF_B2P);
  _Float16* w3t = (_Float16*)(ws + OFF_W3T);
  _Float16* pe  = (_Float16*)(ws + OFF_PE);

  prep_posenc<<<8192, 256, 0, stream>>>(x, pe);

  prep_wv<<<1024, 64, 0, stream>>>(latent,
                                   pw0, pb0, lnw0, lnb0,
                                   pw1, pb1, lnw1, lnb1,
                                   pw2, pb2, lnw2, lnb2,
                                   pw3, pb3, lnw3, lnb3,
                                   B0, B1, B2, B3);

  int mlp_total = N_W1T + N_W2T + 224 + N_W3T;
  prep_mlp<<<(mlp_total + 255) / 256, 256, 0, stream>>>(w1, b1, w2, b2, w3,
                                                        w1t, w2t, b1p, b2p, w3t);

  sdf_main<<<2048, 128, 0, stream>>>(pe, mb0, mb1, mb2, mb3, b3,
                                     B0, B1, B2, B3, w1t, b1p, w2t, b2p, w3t,
                                     (float*)d_out);
}
